// WOQ_Linear_Asymmetric_Model_15668040696350
// MI455X (gfx1250) — compile-verified, matrix operand order checked by reading, not run on hardware
//
#include <hip/hip_runtime.h>

// WOQ int4 (tinygemm asymmetric) linear for MI455X (gfx1250, wave32, WMMA).
//
// out[64x8192] = x @ W^T,  W = (q - 8)*scale_g + zero_g  (group=128 along K)
//
// Weight-streaming bound: 256 MB of int32 q values -> ~11 us floor @ 23.3 TB/s.
// Factorization per group g:  sum_k x*((q-8)s+z) = s*sum_k x*(q-8) + z*sum_k x
//   -> WMMA runs on exact (q-8) in f16, f32 group accumulators, scale/zero in f32.
//
// Tiling: each wave owns 16 distinct N columns x 32 M rows (two 16x16 tiles),
// so each weight element is loaded + dequantized by two waves (one per M half), each reusing it for
// two WMMAs (A fragments come from LDS). 128 blocks x 8 waves = 1024 waves.

typedef _Float16 v16h __attribute__((ext_vector_type(16)));
typedef _Float16 v8h  __attribute__((ext_vector_type(8)));
typedef _Float16 v4h  __attribute__((ext_vector_type(4)));
typedef float    v8f  __attribute__((ext_vector_type(8)));
typedef float    v4f  __attribute__((ext_vector_type(4)));
typedef float    v2f  __attribute__((ext_vector_type(2)));
typedef int      v4i  __attribute__((ext_vector_type(4)));

#define M_TOK    64
#define IN_F     8192
#define OUT_F    8192
#define GROUP    128
#define NGROUPS  (IN_F / GROUP)   // 64
#define NTILE    64               // N columns per block (4 N-subtiles x 16)
#define LDS_STRIDE 136            // 128 f16 + 8 f16 pad (272 B row: 16B-aligned, 4-bank rotate)

__global__ __launch_bounds__(256)
void woq_int4_wmma_kernel(const float* __restrict__ x,
                          const int*   __restrict__ qw,
                          const float* __restrict__ snz,   // (NGROUPS, OUT_F, 2): [g][n][0]=scale, [1]=zero
                          float*       __restrict__ out)
{
    __shared__ _Float16 lx[M_TOK][LDS_STRIDE];  // x chunk, f16
    __shared__ float    psum[128];              // half-row partial sums
    __shared__ float    xsum[M_TOK];            // per-row sum of x over current group

    const int tid  = threadIdx.x;
    const int wave = tid >> 5;
    const int lane = tid & 31;
    const int hi   = lane >> 4;          // 0: K {0..7,16..23}; 1: K {8..15,24..31}
    const int l16  = lane & 15;
    const int nt   = wave & 3;           // 4 N-subtiles -> distinct weight rows per wave
    const int mh   = wave >> 2;          // M half: rows [0,32) or [32,64)
    const int n    = blockIdx.x * NTILE + nt * 16 + l16;  // this lane's output column
    const int mrow0 = mh * 32 + l16;     // A row, M-tile 0
    const int mrow1 = mrow0 + 16;        // A row, M-tile 1
    const int off0 = hi * 8;

    const int* __restrict__ wrow = qw + (size_t)n * IN_F;

    v8f acc0 = {};  // f32 output accumulators, M-tile 0 (VGPR r -> M = mh*32 + hi*8 + r)
    v8f acc1 = {};  // M-tile 1 (M = mh*32 + 16 + hi*8 + r)

    for (int kc = 0; kc < NGROUPS; ++kc) {
        const int kbase = kc * GROUP;

        __syncthreads();  // protect LDS from previous iteration's readers

        // ---- stage x[:, kbase:kbase+128] into LDS as f16; deterministic row sums ----
        if (tid < 128) {
            const int row  = tid >> 1;
            const int half = tid & 1;
            const float* src = x + (size_t)row * IN_F + kbase + half * 64;
            float s = 0.f;
            #pragma unroll
            for (int i = 0; i < 16; ++i) {
                v4f v = *(const v4f*)(src + i * 4);     // global_load_b128
                s += v[0] + v[1] + v[2] + v[3];
                v4h h;
                h[0] = (_Float16)v[0]; h[1] = (_Float16)v[1];
                h[2] = (_Float16)v[2]; h[3] = (_Float16)v[3];
                *(v4h*)&lx[row][half * 64 + i * 4] = h; // ds_store_b64
            }
            psum[tid] = s;
        }
        __syncthreads();
        if (tid < M_TOK) xsum[tid] = psum[2 * tid] + psum[2 * tid + 1];
        __syncthreads();

        // speculative prefetch of next group's weights -> global_prefetch_b8
        if (kc + 1 < NGROUPS) {
            __builtin_prefetch(wrow + kbase + GROUP, 0, 0);
            __builtin_prefetch(wrow + kbase + GROUP + 64, 0, 0);
        }

        // ---- per-group integer GEMM: cg{0,1} += x_f16 @ (q-8)_f16, K = 128 = 4 x 32 ----
        v8f cg0 = {};
        v8f cg1 = {};
        const int* __restrict__ wp = wrow + kbase;
        #pragma unroll
        for (int k32 = 0; k32 < 4; ++k32) {
            const int kb = k32 * 32;

            // A fragments from LDS (16-bit A 16x32 layout), two M-tiles
            v8h a0lo = *(const v8h*)&lx[mrow0][kb + off0];        // ds_load_b128
            v8h a0hi = *(const v8h*)&lx[mrow0][kb + off0 + 16];
            v8h a1lo = *(const v8h*)&lx[mrow1][kb + off0];
            v8h a1hi = *(const v8h*)&lx[mrow1][kb + off0 + 16];
            v16h a0, a1;
            #pragma unroll
            for (int j = 0; j < 8; ++j) {
                a0[j] = a0lo[j]; a0[8 + j] = a0hi[j];
                a1[j] = a1lo[j]; a1[8 + j] = a1hi[j];
            }

            // B fragment: stream q as int32x4, dequant core (q-8) exactly into f16.
            // Loaded + converted ONCE, used by two WMMAs.
            v4i q0 = *(const v4i*)(wp + kb + off0);               // global_load_b128 x4
            v4i q1 = *(const v4i*)(wp + kb + off0 + 4);
            v4i q2 = *(const v4i*)(wp + kb + off0 + 16);
            v4i q3 = *(const v4i*)(wp + kb + off0 + 20);
            v16h b;
            #pragma unroll
            for (int j = 0; j < 4; ++j) {
                b[j]      = (_Float16)(q0[j] - 8);
                b[4 + j]  = (_Float16)(q1[j] - 8);
                b[8 + j]  = (_Float16)(q2[j] - 8);
                b[12 + j] = (_Float16)(q3[j] - 8);
            }

            cg0 = __builtin_amdgcn_wmma_f32_16x16x32_f16(
                      false, a0, false, b, (short)0, cg0, false, false);
            cg1 = __builtin_amdgcn_wmma_f32_16x16x32_f16(
                      false, a1, false, b, (short)0, cg1, false, false);
        }

        // ---- epilogue: acc += scale_g * cg + zero_g * rowsum_g(x) ----
        v2f sz = *(const v2f*)(snz + ((size_t)kc * OUT_F + n) * 2);  // global_load_b64
        const float s = sz[0];
        const float z = sz[1];
        const int mb = mh * 32 + hi * 8;
        #pragma unroll
        for (int r = 0; r < 8; ++r) {
            float t0 = fmaf(s, cg0[r], acc0[r]);
            acc0[r]  = fmaf(z, xsum[mb + r], t0);
            float t1 = fmaf(s, cg1[r], acc1[r]);
            acc1[r]  = fmaf(z, xsum[mb + 16 + r], t1);
        }
    }

    // ---- store two 16x16 f32 tiles ----
    const int mb = mh * 32 + hi * 8;
    #pragma unroll
    for (int r = 0; r < 8; ++r) {
        out[(size_t)(mb + r) * OUT_F + n]      = acc0[r];
        out[(size_t)(mb + 16 + r) * OUT_F + n] = acc1[r];
    }
}

extern "C" void kernel_launch(void* const* d_in, const int* in_sizes, int n_in,
                              void* d_out, int out_size, void* d_ws, size_t ws_size,
                              hipStream_t stream) {
    (void)in_sizes; (void)n_in; (void)out_size; (void)d_ws; (void)ws_size;
    const float* x   = (const float*)d_in[0];   // (64, 8192) f32
    const int*   qw  = (const int*)d_in[1];     // (8192, 8192) int32 in [0,15]
    const float* snz = (const float*)d_in[2];   // (64, 8192, 2) f32
    float* out = (float*)d_out;                 // (64, 8192) f32

    dim3 grid(OUT_F / NTILE);   // 128 blocks
    dim3 block(256);            // 8 wave32 waves
    hipLaunchKernelGGL(woq_int4_wmma_kernel, grid, block, 0, stream, x, qw, snz, out);
}
